// TTMiniMaxM1LightningAttention_1657857376611
// MI455X (gfx1250) — matrix operand order repaired and checked by reading, not compiled, not run on hardware
//
#include <hip/hip_runtime.h>
#include <hip/hip_bf16.h>

// ---------------------------------------------------------------------------
// MiniMax-M1 Lightning Attention for MI455X (gfx1250, wave32, WMMA bf16)
//   1) cvt: fp32 -> bf16 packs of X, [Wqkv;Wgate], Wout
//   2) fused GEMM (8192x8192x2048), async-LDS double-buffered, silu/sigmoid
//   3) lightning attention scan, all einsums via v_wmma_f32_16x16x32_bf16
//   4) rmsnorm * gate
//   5) output GEMM (8192x2048x2048) -> fp32 d_out
// ---------------------------------------------------------------------------

typedef __bf16 bf16;
typedef __attribute__((ext_vector_type(16))) __bf16 v16bf;
typedef __attribute__((ext_vector_type(8)))  __bf16 v8bf;
typedef __attribute__((ext_vector_type(8)))  float  v8f;

#define B_    2
#define NSEQ  4096
#define HID   2048
#define NH    16
#define DH    128
#define BLK   256
#define NB    16
#define MTOT  8192
#define NQKV  6144
#define NFUSE 8192

// LDS tile geometry for the GEMMs (bf16 elements)
#define TSTRIDE 72            // 144B row stride: 16B aligned, conflict-free
#define ATILE   (128 * TSTRIDE)
#define STAGE   (2 * ATILE)   // A tile + B tile per stage

__device__ __forceinline__ float sigmoidf_(float x) { return 1.f / (1.f + __expf(-x)); }

// ---- CDNA5 async global->LDS copy (ASYNCcnt) ------------------------------
// LDS byte address = low 32 bits of the generic shared-aperture pointer.
__device__ __forceinline__ void async_ld_b128(const void* gptr, void* lptr) {
  uint32_t lds_addr = (uint32_t)(uintptr_t)lptr;
  asm volatile("global_load_async_to_lds_b128 %0, %1, off"
               :: "v"(lds_addr), "v"((uint64_t)(uintptr_t)gptr)
               : "memory");
}
__device__ __forceinline__ void wait_async0() {
  asm volatile("s_wait_asynccnt 0x0" ::: "memory");
}

// ---- WMMA fragment loaders (bf16, 16x16x32) -------------------------------
__device__ __forceinline__ v16bf load_frag_a(const bf16* __restrict__ src, int ld,
                                             int row0, int k0, int lane) {
  int m  = row0 + (lane & 15);
  int kh = (lane >> 4) * 8;
  const bf16* p = src + (size_t)m * ld + (k0 + kh);
  v8bf lo = *(const v8bf*)(p);        // K = kh .. kh+7
  v8bf hi = *(const v8bf*)(p + 16);   // K = kh+16 .. kh+23
  v16bf r;
#pragma unroll
  for (int i = 0; i < 8; ++i) { r[i] = lo[i]; r[i + 8] = hi[i]; }
  return r;
}

__device__ __forceinline__ v16bf load_frag_b(const bf16* __restrict__ src, int ld,
                                             int n0, int k0, int lane) {
  int n  = n0 + (lane & 15);
  int ks = k0 + (lane >> 4) * 8;
  const bf16* p = src + (size_t)n * ld + ks;
  v8bf lo = *(const v8bf*)(p);
  v8bf hi = *(const v8bf*)(p + 16);
  v16bf r;
#pragma unroll
  for (int i = 0; i < 8; ++i) { r[i] = lo[i]; r[i + 8] = hi[i]; }
  return r;
}

__device__ __forceinline__ v8f wmma_bf16(v16bf a, v16bf b, v8f c) {
  return __builtin_amdgcn_wmma_f32_16x16x32_bf16(false, a, false, b, (short)0, c,
                                                 false, false);
}

// ---- fp32 -> bf16 conversion ----------------------------------------------
__global__ void cvt_f32_bf16(const float* __restrict__ s, bf16* __restrict__ d, int n) {
  int i = blockIdx.x * blockDim.x + threadIdx.x;
  if (i < n) d[i] = (bf16)s[i];
}

// ---- tiled bf16 GEMM:  C[m,n] = sum_k A[m,k] * W[n,k] ---------------------
// 256 threads = 8 waves; tile 128(M) x 128(N); wave -> 64x32 subtile.
// K staged 64-wide into double-buffered LDS via async global->LDS copies.
// MODE 0: fused qkv+gate epilogue; MODE 1: plain fp32 store.
template <int MODE>
__global__ __launch_bounds__(256) void gemm_bf16_kernel(
    const bf16* __restrict__ A, const bf16* __restrict__ W,
    int N, int Kdim,
    bf16* __restrict__ Qo, bf16* __restrict__ Ko, bf16* __restrict__ Vo,
    bf16* __restrict__ Gateo, float* __restrict__ Co) {
  extern __shared__ bf16 lds[];  // [2][ A:128*72 | B:128*72 ]
  const int lane = threadIdx.x & 31;
  const int wave = threadIdx.x >> 5;
  const int wm = wave >> 2;  // 0..1
  const int wn = wave & 3;   // 0..3
  const int rowbase = blockIdx.y * 128;
  const int colbase = blockIdx.x * 128;

  auto stage = [&](int buf, int k0) {
    bf16* dA = lds + buf * STAGE;
    bf16* dB = dA + ATILE;
    const bf16* gA = A + (size_t)rowbase * Kdim + k0;
    const bf16* gB = W + (size_t)colbase * Kdim + k0;
#pragma unroll
    for (int i = 0; i < 4; ++i) {
      int c = threadIdx.x + i * 256;  // 0..1023: 128 rows x 8 chunks of 16B
      int row = c >> 3;
      int kk = (c & 7) * 8;
      async_ld_b128(gA + (size_t)row * Kdim + kk, dA + row * TSTRIDE + kk);
      async_ld_b128(gB + (size_t)row * Kdim + kk, dB + row * TSTRIDE + kk);
    }
  };

  v8f acc[4][2];
#pragma unroll
  for (int i = 0; i < 4; ++i)
#pragma unroll
    for (int j = 0; j < 2; ++j) acc[i][j] = {};

  stage(0, 0);
  wait_async0();
  __syncthreads();

  const int KT = Kdim / 64;
  int buf = 0;
  for (int kt = 0; kt < KT; ++kt) {
    if (kt + 1 < KT) stage(buf ^ 1, (kt + 1) * 64);  // overlap with compute
    const bf16* sA = lds + buf * STAGE;
    const bf16* sB = sA + ATILE;
#pragma unroll
    for (int kc = 0; kc < 64; kc += 32) {
      v16bf a[4], b[2];
#pragma unroll
      for (int i = 0; i < 4; ++i)
        a[i] = load_frag_a(sA, TSTRIDE, wm * 64 + 16 * i, kc, lane);
#pragma unroll
      for (int j = 0; j < 2; ++j)
        b[j] = load_frag_b(sB, TSTRIDE, wn * 32 + 16 * j, kc, lane);
#pragma unroll
      for (int i = 0; i < 4; ++i)
#pragma unroll
        for (int j = 0; j < 2; ++j) acc[i][j] = wmma_bf16(a[i], b[j], acc[i][j]);
    }
    wait_async0();
    __syncthreads();
    buf ^= 1;
  }

  // C layout: lane&15 -> n, vgpr r -> m = r + 8*(lane>>4)
#pragma unroll
  for (int i = 0; i < 4; ++i) {
#pragma unroll
    for (int j = 0; j < 2; ++j) {
      int c = colbase + wn * 32 + 16 * j + (lane & 15);
      int mrow0 = rowbase + wm * 64 + 16 * i + ((lane >> 4) * 8);
#pragma unroll
      for (int r = 0; r < 8; ++r) {
        int m = mrow0 + r;
        float v = acc[i][j][r];
        if (MODE == 0) {
          if (c < NQKV) {
            float sv = v * sigmoidf_(v);  // silu
            int which = c >> 11;          // 0=q 1=k 2=v
            int rem = c & 2047;
            int head = rem >> 7;
            int dd = rem & 127;
            int bidx = m >> 12;
            int seq = m & 4095;
            size_t di = ((size_t)(bidx * NH + head) * NSEQ + seq) * DH + dd;
            bf16 bv = (bf16)sv;
            if (which == 0)
              Qo[di] = bv;
            else if (which == 1)
              Ko[di] = bv;
            else
              Vo[di] = bv;
          } else {
            Gateo[(size_t)m * HID + (c - NQKV)] = (bf16)sigmoidf_(v);
          }
        } else {
          Co[(size_t)m * N + c] = v;
        }
      }
    }
  }
}

// ---- lightning attention scan ---------------------------------------------
// One workgroup (256 thr, 8 waves) per (b,h). Sequential over 16 blocks.
// LDS (305152 B dynamic, within the 320KB WGP budget).
__global__ __launch_bounds__(256) void lightning_attn_kernel(
    const bf16* __restrict__ Q, const bf16* __restrict__ K,
    const bf16* __restrict__ V, const float* __restrict__ slope,
    float* __restrict__ attn, float* __restrict__ kv_ws) {
  extern __shared__ bf16 smem[];
  bf16* qk_s = smem;               // 256*264
  bf16* kT_s = qk_s + 256 * 264;   // 128*264
  bf16* vT_s = kT_s + 128 * 264;   // 128*264
  bf16* kvT_s = vT_s + 128 * 264;  // 128*136

  const int lane = threadIdx.x & 31;
  const int wave = threadIdx.x >> 5;
  const int t = threadIdx.x;
  const int wg = blockIdx.x;
  const int bidx = wg >> 4;
  const int head = wg & 15;
  const float s = slope[head];
  const float bdecay = __expf(-s * (float)BLK);

  const bf16* Qh = Q + (size_t)(bidx * NH + head) * NSEQ * DH;
  const bf16* Kh = K + (size_t)(bidx * NH + head) * NSEQ * DH;
  const bf16* Vh = V + (size_t)(bidx * NH + head) * NSEQ * DH;
  float* kv = kv_ws + (size_t)wg * DH * DH;

  for (int i = t; i < DH * DH; i += 256) kv[i] = 0.f;
  __threadfence_block();

  for (int bi = 0; bi < NB; ++bi) {
    __syncthreads();

    for (int i = t; i < DH * DH; i += 256) {
      int d = i >> 7, e = i & 127;
      kvT_s[e * 136 + d] = (bf16)kv[i];
    }
    const bf16* Qb = Qh + (size_t)bi * BLK * DH;
    const bf16* Kb = Kh + (size_t)bi * BLK * DH;
    const bf16* Vb = Vh + (size_t)bi * BLK * DH;

    for (int i = t; i < BLK * DH; i += 256) {
      int seq = i >> 7, d = i & 127;
      float kd = __expf(-s * (float)(BLK - 1 - seq));
      kT_s[d * 264 + seq] = (bf16)((float)Kb[i] * kd);
      vT_s[d * 264 + seq] = Vb[i];
    }
    __syncthreads();

    // ---- qk = (Qb @ Kb^T) * diag_decay -> qk_s (bf16) ----
    {
      const int m0w = wave * 32;
      v16bf aq[2][4];
#pragma unroll
      for (int rt = 0; rt < 2; ++rt)
#pragma unroll
        for (int kk = 0; kk < 4; ++kk)
          aq[rt][kk] = load_frag_a(Qb, DH, m0w + rt * 16, kk * 32, lane);
#pragma unroll
      for (int rt = 0; rt < 2; ++rt) {
        int m0 = m0w + rt * 16;
        for (int ct = 0; ct < 16; ++ct) {
          int j0 = ct * 16;
          v8f acc = {};
#pragma unroll
          for (int kk = 0; kk < 4; ++kk) {
            v16bf b = load_frag_b(Kb, DH, j0, kk * 32, lane);
            acc = wmma_bf16(aq[rt][kk], b, acc);
          }
          int j = j0 + (lane & 15);
          int mb = m0 + ((lane >> 4) * 8);
#pragma unroll
          for (int r = 0; r < 8; ++r) {
            int m = mb + r;
            int diff = m - j;
            float dv = (diff >= 0) ? __expf(-s * (float)diff) : 0.f;
            qk_s[m * 264 + j] = (bf16)(acc[r] * dv);
          }
        }
      }
    }
    __syncthreads();

    // ---- out = (q*q_decay) @ kv  +  qk @ v ----
    {
      const int m0w = wave * 32;
#pragma unroll
      for (int rt = 0; rt < 2; ++rt) {
        int m0 = m0w + rt * 16;
        float fq = __expf(-s * (float)(m0 + (lane & 15) + 1));
        v16bf qd[4];
#pragma unroll
        for (int kk = 0; kk < 4; ++kk) {
          v16bf a = load_frag_a(Qb, DH, m0, kk * 32, lane);
#pragma unroll
          for (int x = 0; x < 16; ++x) a[x] = (bf16)((float)a[x] * fq);
          qd[kk] = a;
        }
        v16bf aqk[8];
#pragma unroll
        for (int kk = 0; kk < 8; ++kk)
          aqk[kk] = load_frag_a(qk_s, 264, m0, kk * 32, lane);
        for (int et = 0; et < 8; ++et) {
          int e0 = et * 16;
          v8f acc = {};
#pragma unroll
          for (int kk = 0; kk < 4; ++kk) {
            v16bf b = load_frag_b(kvT_s, 136, e0, kk * 32, lane);
            acc = wmma_bf16(qd[kk], b, acc);
          }
#pragma unroll
          for (int kk = 0; kk < 8; ++kk) {
            v16bf b = load_frag_b(vT_s, 264, e0, kk * 32, lane);
            acc = wmma_bf16(aqk[kk], b, acc);
          }
          int e = e0 + (lane & 15);
          int mb = m0 + ((lane >> 4) * 8);
#pragma unroll
          for (int r = 0; r < 8; ++r) {
            int seq = bi * BLK + mb + r;
            attn[((size_t)bidx * NSEQ + seq) * HID + head * DH + e] = acc[r];
          }
        }
      }
    }

    // ---- kv = block_decay*kv + (k*k_decay)^T @ v ----
    {
      int d0 = wave * 16;
      v16bf at[8];
#pragma unroll
      for (int kk = 0; kk < 8; ++kk)
        at[kk] = load_frag_a(kT_s, 264, d0, kk * 32, lane);
      for (int et = 0; et < 8; ++et) {
        int e0 = et * 16;
        v8f acc = {};
#pragma unroll
        for (int kk = 0; kk < 8; ++kk) {
          v16bf b = load_frag_b(vT_s, 264, e0, kk * 32, lane);
          acc = wmma_bf16(at[kk], b, acc);
        }
        int e = e0 + (lane & 15);
        int db = d0 + ((lane >> 4) * 8);
#pragma unroll
        for (int r = 0; r < 8; ++r) {
          int d = db + r;
          float old = kv[d * DH + e];
          kv[d * DH + e] = bdecay * old + acc[r];
        }
      }
    }
    __threadfence_block();
  }
}

// ---- rmsnorm * gate -> bf16 ------------------------------------------------
__global__ __launch_bounds__(256) void norm_gate_kernel(
    const float* __restrict__ attn, const bf16* __restrict__ gate,
    const float* __restrict__ normw, bf16* __restrict__ y) {
  __shared__ float red[256];
  int m = blockIdx.x;
  int t = threadIdx.x;
  const float* row = attn + (size_t)m * HID;
  float xv[8];
  float ss = 0.f;
#pragma unroll
  for (int i = 0; i < 8; ++i) {
    xv[i] = row[t + i * 256];
    ss += xv[i] * xv[i];
  }
  red[t] = ss;
  __syncthreads();
  for (int off = 128; off > 0; off >>= 1) {
    if (t < off) red[t] += red[t + off];
    __syncthreads();
  }
  float scale = rsqrtf(red[0] / (float)HID + 1e-6f);
#pragma unroll
  for (int i = 0; i < 8; ++i) {
    int c = t + i * 256;
    float g = (float)gate[(size_t)m * HID + c];
    y[(size_t)m * HID + c] = (bf16)(g * xv[i] * scale * normw[c]);
  }
}

// ---------------------------------------------------------------------------
extern "C" void kernel_launch(void* const* d_in, const int* in_sizes, int n_in,
                              void* d_out, int out_size, void* d_ws, size_t ws_size,
                              hipStream_t stream) {
  const float* hidden = (const float*)d_in[0];
  const float* slope  = (const float*)d_in[1];
  const float* wqkv   = (const float*)d_in[2];
  const float* wgate  = (const float*)d_in[3];
  const float* wout   = (const float*)d_in[4];
  const float* normw  = (const float*)d_in[5];
  (void)in_sizes; (void)n_in; (void)out_size; (void)ws_size;

  char* ws = (char*)d_ws;
  size_t off = 0;
  auto alloc = [&](size_t bytes) -> char* {
    char* p = ws + off;
    off += (bytes + 255) & ~(size_t)255;
    return p;
  };
  bf16* Xbf   = (bf16*)alloc((size_t)MTOT * HID * 2);
  bf16* Wbf   = (bf16*)alloc((size_t)NFUSE * HID * 2);  // [wqkv; wgate]
  bf16* Woutb = (bf16*)alloc((size_t)HID * HID * 2);
  bf16* Qbf   = (bf16*)alloc((size_t)MTOT * HID * 2);
  bf16* Kbf   = (bf16*)alloc((size_t)MTOT * HID * 2);
  bf16* Vbf   = (bf16*)alloc((size_t)MTOT * HID * 2);
  bf16* Gateb = (bf16*)alloc((size_t)MTOT * HID * 2);
  float* attnf = (float*)alloc((size_t)MTOT * HID * 4);
  bf16* Ybf   = (bf16*)alloc((size_t)MTOT * HID * 2);
  float* kvws = (float*)alloc((size_t)B_ * NH * DH * DH * 4);

  // 1) fp32 -> bf16 packs
  cvt_f32_bf16<<<(MTOT * HID + 255) / 256, 256, 0, stream>>>(hidden, Xbf, MTOT * HID);
  cvt_f32_bf16<<<(NQKV * HID + 255) / 256, 256, 0, stream>>>(wqkv, Wbf, NQKV * HID);
  cvt_f32_bf16<<<(HID * HID + 255) / 256, 256, 0, stream>>>(
      wgate, Wbf + (size_t)NQKV * HID, HID * HID);
  cvt_f32_bf16<<<(HID * HID + 255) / 256, 256, 0, stream>>>(wout, Woutb, HID * HID);

  size_t gemm_smem = (size_t)(2 * STAGE) * sizeof(bf16);  // 73728 B

  // 2) fused qkv+gate GEMM with silu/sigmoid epilogue
  gemm_bf16_kernel<0><<<dim3(NFUSE / 128, MTOT / 128), 256, gemm_smem, stream>>>(
      Xbf, Wbf, NFUSE, HID, Qbf, Kbf, Vbf, Gateb, nullptr);

  // 3) lightning attention scan (1 WG per (b,h))
  size_t attn_smem =
      (size_t)(256 * 264 + 128 * 264 + 128 * 264 + 128 * 136) * sizeof(bf16);
  lightning_attn_kernel<<<B_ * NH, 256, attn_smem, stream>>>(Qbf, Kbf, Vbf, slope,
                                                             attnf, kvws);

  // 4) rmsnorm * gate
  norm_gate_kernel<<<MTOT, 256, 0, stream>>>(attnf, Gateb, normw, Ybf);

  // 5) output GEMM -> fp32 d_out
  gemm_bf16_kernel<1><<<dim3(HID / 128, MTOT / 128), 256, gemm_smem, stream>>>(
      Ybf, Woutb, HID, HID, nullptr, nullptr, nullptr, nullptr, (float*)d_out);
}
